// AttentionWithKVCache_81398220194653
// MI455X (gfx1250) — matrix operand order repaired and checked by reading, not compiled, not run on hardware
//
#include <hip/hip_runtime.h>

#define BB   2
#define SS   4096
#define DIMM 1024
#define HH   16
#define KVH  2
#define HD   64
#define HWIN 128

typedef __attribute__((ext_vector_type(16))) __bf16 v16bf;
typedef __attribute__((ext_vector_type(8)))  float  v8f;

union Frag {
  v16bf v;
  uint4 q[2];
  unsigned short u[16];
};

__device__ __forceinline__ unsigned short f2bf(float f) {
  union { float f; unsigned int u; } c; c.f = f;
  return (unsigned short)((c.u + 0x7FFFu + ((c.u >> 16) & 1u)) >> 16);
}

__device__ __forceinline__ int iclamp(int v, int lo, int hi) {
  return v < lo ? lo : (v > hi ? hi : v);
}

#define WMMA_BF16(A, B, C)                                                    \
  __builtin_amdgcn_wmma_f32_16x16x32_bf16(false, (A).v, false, (B).v,         \
                                          (short)0, (C), false, false)

// ---------------------------------------------------------------------------
// Kernel 1: convert x -> bf16, transpose/pack weights to [N][K] bf16,
// concatenate qkv bias.
// ---------------------------------------------------------------------------
__global__ void pack_kernel(const float* __restrict__ x,
                            const float* __restrict__ Wq, const float* __restrict__ bq,
                            const float* __restrict__ Wk, const float* __restrict__ bk,
                            const float* __restrict__ Wv, const float* __restrict__ bv,
                            const float* __restrict__ Wo,
                            unsigned short* __restrict__ xb,
                            unsigned short* __restrict__ WqkvT,
                            unsigned short* __restrict__ WoT,
                            float* __restrict__ biasC) {
  const long stride = (long)gridDim.x * blockDim.x;
  const long tid = (long)blockIdx.x * blockDim.x + threadIdx.x;
  // x: vectorized 4-wide convert
  for (long i = tid; i < (long)BB * SS * DIMM / 4; i += stride) {
    float4 f = ((const float4*)x)[i];
    ushort4 r;
    r.x = f2bf(f.x); r.y = f2bf(f.y); r.z = f2bf(f.z); r.w = f2bf(f.w);
    ((ushort4*)xb)[i] = r;
  }
  for (long i = tid; i < 1280L * DIMM; i += stride) {
    int n = (int)(i >> 10), k = (int)(i & 1023);
    float w;
    if (n < 1024)      w = Wq[(long)k * 1024 + n];
    else if (n < 1152) w = Wk[(long)k * 128 + (n - 1024)];
    else               w = Wv[(long)k * 128 + (n - 1152)];
    WqkvT[i] = f2bf(w);
  }
  for (long i = tid; i < 1024L * 1024L; i += stride) {
    int n = (int)(i >> 10), k = (int)(i & 1023);
    WoT[i] = f2bf(Wo[(long)k * 1024 + n]);
  }
  for (long i = tid; i < 1280; i += stride)
    biasC[i] = (i < 1024) ? bq[i] : ((i < 1152) ? bk[i - 1024] : bv[i - 1152]);
}

// ---------------------------------------------------------------------------
// Shared GEMM machinery: per-wave 32Mx64N tile, K-loop step 32, double
// buffered so next step's global loads overlap current step's 8 WMMAs.
// A is [M][K] bf16 row-major, B is [N][K] bf16 row-major (pre-transposed).
// ---------------------------------------------------------------------------
struct GemmBufs { Frag a0, a1, b0, b1, b2, b3; };

__device__ __forceinline__ void gemm_load(GemmBufs& t,
                                          const unsigned short* __restrict__ A0,
                                          const unsigned short* __restrict__ A1,
                                          const unsigned short* __restrict__ B0,
                                          const unsigned short* __restrict__ B1,
                                          const unsigned short* __restrict__ B2,
                                          const unsigned short* __restrict__ B3,
                                          int k0, int half) {
  t.a0.q[0] = *(const uint4*)(A0 + k0 + 16 * half);
  t.a0.q[1] = *(const uint4*)(A0 + k0 + 16 * half + 8);
  t.a1.q[0] = *(const uint4*)(A1 + k0 + 16 * half);
  t.a1.q[1] = *(const uint4*)(A1 + k0 + 16 * half + 8);
  t.b0.q[0] = *(const uint4*)(B0 + k0);
  t.b0.q[1] = *(const uint4*)(B0 + k0 + 8);
  t.b1.q[0] = *(const uint4*)(B1 + k0);
  t.b1.q[1] = *(const uint4*)(B1 + k0 + 8);
  t.b2.q[0] = *(const uint4*)(B2 + k0);
  t.b2.q[1] = *(const uint4*)(B2 + k0 + 8);
  t.b3.q[0] = *(const uint4*)(B3 + k0);
  t.b3.q[1] = *(const uint4*)(B3 + k0 + 8);
}

__device__ __forceinline__ void gemm_mma(v8f acc[8], const GemmBufs& t) {
  acc[0] = WMMA_BF16(t.a0, t.b0, acc[0]);
  acc[1] = WMMA_BF16(t.a0, t.b1, acc[1]);
  acc[2] = WMMA_BF16(t.a0, t.b2, acc[2]);
  acc[3] = WMMA_BF16(t.a0, t.b3, acc[3]);
  acc[4] = WMMA_BF16(t.a1, t.b0, acc[4]);
  acc[5] = WMMA_BF16(t.a1, t.b1, acc[5]);
  acc[6] = WMMA_BF16(t.a1, t.b2, acc[6]);
  acc[7] = WMMA_BF16(t.a1, t.b3, acc[7]);
}

// Computes acc[8] = A(32xK) @ B^T-layout(64xK) for one wave's tile.
__device__ __forceinline__ void gemm_core(v8f acc[8],
                                          const unsigned short* __restrict__ Ab,
                                          const unsigned short* __restrict__ WT,
                                          int mBase, int nBase, int half, int ln) {
#pragma unroll
  for (int i = 0; i < 8; ++i)
#pragma unroll
    for (int j = 0; j < 8; ++j) acc[i][j] = 0.f;
  const unsigned short* A0 = Ab + (long)(mBase + ln) * DIMM;
  const unsigned short* A1 = Ab + (long)(mBase + 16 + ln) * DIMM;
  const unsigned short* B0 = WT + (long)(nBase + 0 * 16 + ln) * DIMM + 16 * half;
  const unsigned short* B1 = WT + (long)(nBase + 1 * 16 + ln) * DIMM + 16 * half;
  const unsigned short* B2 = WT + (long)(nBase + 2 * 16 + ln) * DIMM + 16 * half;
  const unsigned short* B3 = WT + (long)(nBase + 3 * 16 + ln) * DIMM + 16 * half;

  GemmBufs bufA, bufB;
  gemm_load(bufA, A0, A1, B0, B1, B2, B3, 0, half);
  for (int k0 = 0; k0 < DIMM; k0 += 64) {
    gemm_load(bufB, A0, A1, B0, B1, B2, B3, k0 + 32, half);
    gemm_mma(acc, bufA);
    if (k0 + 64 < DIMM)
      gemm_load(bufA, A0, A1, B0, B1, B2, B3, k0 + 64, half);
    gemm_mma(acc, bufB);
  }
}

// ---------------------------------------------------------------------------
// Kernel 2: qkv = x @ [Wq|Wk|Wv] + bias.  M=8192, N=1280, K=1024.
// Scatter epilogue: Q -> [b][h][s][d], K -> [b][kv][s][d],
// V -> [b][kv][d][s] (transposed so attention PV B-fragments are contiguous).
// ---------------------------------------------------------------------------
__global__ __launch_bounds__(256) void qkv_gemm_kernel(
    const unsigned short* __restrict__ xb, const unsigned short* __restrict__ WT,
    const float* __restrict__ biasC,
    unsigned short* __restrict__ Qb, unsigned short* __restrict__ Kb,
    unsigned short* __restrict__ Vtb) {
  const int wave = threadIdx.x >> 5, lane = threadIdx.x & 31;
  const int half = lane >> 4, ln = lane & 15;
  const int mBase = (blockIdx.x * 8 + wave) * 32;
  const int nBase = blockIdx.y * 64;
  v8f acc[8];
  gemm_core(acc, xb, WT, mBase, nBase, half, ln);

#pragma unroll
  for (int sub = 0; sub < 2; ++sub) {
#pragma unroll
    for (int c = 0; c < 4; ++c) {
#pragma unroll
      for (int i = 0; i < 8; ++i) {
        const int m = mBase + sub * 16 + i + 8 * half;
        const int n = nBase + c * 16 + ln;
        const unsigned short bvv = f2bf(acc[sub * 4 + c][i] + biasC[n]);
        const int b = m >> 12, s = m & (SS - 1);
        const int d = n & 63;
        if (n < 1024) {
          const int h = n >> 6;
          Qb[((long)(b * HH + h) * SS + s) * HD + d] = bvv;
        } else if (n < 1152) {
          const int kv = (n - 1024) >> 6;
          Kb[((long)(b * KVH + kv) * SS + s) * HD + d] = bvv;
        } else {
          const int kv = (n - 1152) >> 6;
          Vtb[((long)(b * KVH + kv) * HD + d) * SS + s] = bvv;
        }
      }
    }
  }
}

// ---------------------------------------------------------------------------
// Kernel 3: sliding-window GQA attention, one wave per 16-query tile.
// Online softmax over 9 steps of 32 keys covering [q0-128, q0+159].
// Zero-padded boundary keys get raw score 0 (participate in softmax) and
// contribute no value (P forced to 0 in the LDS staging buffer).
// ---------------------------------------------------------------------------
__global__ __launch_bounds__(256) void attn_kernel(
    const unsigned short* __restrict__ Qb, const unsigned short* __restrict__ Kb,
    const unsigned short* __restrict__ Vtb, unsigned short* __restrict__ attnO) {
  __shared__ unsigned short ldsP[8][16][32];
  const int wave = threadIdx.x >> 5, lane = threadIdx.x & 31;
  const int half = lane >> 4, ln = lane & 15;
  const int gw = blockIdx.x * 8 + wave;
  const int qt = gw & 255;
  const int bh = gw >> 8;
  const int h = bh & (HH - 1), b = bh >> 4;
  const int kv = h >> 3;  // H/KV = 8 query heads per kv head
  const int q0 = qt * 16;
  const unsigned short* Qp = Qb + ((long)(b * HH + h) * SS + q0) * HD;
  const unsigned short* Kp = Kb + ((long)(b * KVH + kv) * SS) * HD;
  const unsigned short* Vp = Vtb + ((long)(b * KVH + kv) * HD) * SS;

  // A-fragments of Q (16x32 bf16 x2 covering d=0..63), loaded once.
  Frag aQ0, aQ1;
  {
    const unsigned short* r = Qp + ln * HD;
    aQ0.q[0] = *(const uint4*)(r + 16 * half);
    aQ0.q[1] = *(const uint4*)(r + 16 * half + 8);
    aQ1.q[0] = *(const uint4*)(r + 32 + 16 * half);
    aQ1.q[1] = *(const uint4*)(r + 40 + 16 * half);
  }

  float mrow[8], lrow[8];
  v8f o[4];
#pragma unroll
  for (int i = 0; i < 8; ++i) { mrow[i] = -1e30f; lrow[i] = 0.f; }
#pragma unroll
  for (int c = 0; c < 4; ++c)
#pragma unroll
    for (int i = 0; i < 8; ++i) o[c][i] = 0.f;

  for (int step = 0; step < 9; ++step) {
    const int kBase = q0 - HWIN + step * 32;
    v8f s0, s1;
#pragma unroll
    for (int i = 0; i < 8; ++i) { s0[i] = 0.f; s1[i] = 0.f; }

    // scores for keys [kBase, kBase+16) and [kBase+16, kBase+32)
    {
      const int kc = iclamp(kBase + ln, 0, SS - 1);
      const unsigned short* r = Kp + (long)kc * HD;
      Frag b0, b1;
      b0.q[0] = *(const uint4*)(r + 16 * half);
      b0.q[1] = *(const uint4*)(r + 16 * half + 8);
      b1.q[0] = *(const uint4*)(r + 32 + 16 * half);
      b1.q[1] = *(const uint4*)(r + 40 + 16 * half);
      s0 = WMMA_BF16(aQ0, b0, s0);
      s0 = WMMA_BF16(aQ1, b1, s0);
    }
    {
      const int kc = iclamp(kBase + 16 + ln, 0, SS - 1);
      const unsigned short* r = Kp + (long)kc * HD;
      Frag b0, b1;
      b0.q[0] = *(const uint4*)(r + 16 * half);
      b0.q[1] = *(const uint4*)(r + 16 * half + 8);
      b1.q[0] = *(const uint4*)(r + 32 + 16 * half);
      b1.q[1] = *(const uint4*)(r + 40 + 16 * half);
      s1 = WMMA_BF16(aQ0, b0, s1);
      s1 = WMMA_BF16(aQ1, b1, s1);
    }

    // Issue the V^T B-fragment loads NOW so their latency hides under the
    // softmax VALU work and LDS staging below.
    Frag vf[4];
    {
      const int gBase = iclamp(kBase + 16 * half, 0, SS - 16);
#pragma unroll
      for (int c = 0; c < 4; ++c) {
        const unsigned short* p = Vp + (long)(c * 16 + ln) * SS + gBase;
        vf[c].q[0] = *(const uint4*)p;
        vf[c].q[1] = *(const uint4*)(p + 8);
      }
    }

    // Mask (C layout: lane -> N/key, VGPR i -> M/query row i + 8*half).
    const int key0 = kBase + ln;
    const int key1 = key0 + 16;
    const bool v0 = (key0 >= 0) && (key0 < SS);
    const bool v1 = (key1 >= 0) && (key1 < SS);
#pragma unroll
    for (int i = 0; i < 8; ++i) {
      const int qrow = q0 + i + 8 * half;
      const bool w0 = (key0 >= qrow - HWIN) && (key0 < qrow + HWIN);
      const bool w1 = (key1 >= qrow - HWIN) && (key1 < qrow + HWIN);
      s0[i] = w0 ? (v0 ? s0[i] : 0.f) : -1e30f;
      s1[i] = w1 ? (v1 ? s1[i] : 0.f) : -1e30f;
    }

    // Online softmax: row max / rescale / exp / row sum.
    float mnew[8], scale[8], p0[8], p1[8];
#pragma unroll
    for (int i = 0; i < 8; ++i) {
      float t = fmaxf(s0[i], s1[i]);
#pragma unroll
      for (int xm = 1; xm < 16; xm <<= 1) t = fmaxf(t, __shfl_xor(t, xm, 32));
      mnew[i] = fmaxf(mrow[i], t);
      scale[i] = __expf(mrow[i] - mnew[i]);
      mrow[i] = mnew[i];
    }
#pragma unroll
    for (int c = 0; c < 4; ++c)
#pragma unroll
      for (int i = 0; i < 8; ++i) o[c][i] *= scale[i];
#pragma unroll
    for (int i = 0; i < 8; ++i) {
      p0[i] = __expf(s0[i] - mnew[i]);
      p1[i] = __expf(s1[i] - mnew[i]);
      float r = p0[i] + p1[i];
#pragma unroll
      for (int xm = 1; xm < 16; xm <<= 1) r += __shfl_xor(r, xm, 32);
      lrow[i] = lrow[i] * scale[i] + r;
    }

    // Stage P (16x32 bf16) in LDS; zero out padded keys so garbage V is inert.
#pragma unroll
    for (int i = 0; i < 8; ++i) {
      ldsP[wave][i + 8 * half][ln] = v0 ? f2bf(p0[i]) : (unsigned short)0;
      ldsP[wave][i + 8 * half][16 + ln] = v1 ? f2bf(p1[i]) : (unsigned short)0;
    }
    asm volatile("" ::: "memory");  // keep ds reads after ds writes (LDS in-order)

    // P as A-fragment (layout conversion through LDS), then 4 PV WMMAs.
    Frag aP;
    aP.q[0] = *(const uint4*)(&ldsP[wave][ln][16 * half]);
    aP.q[1] = *(const uint4*)(&ldsP[wave][ln][16 * half + 8]);
#pragma unroll
    for (int c = 0; c < 4; ++c) o[c] = WMMA_BF16(aP, vf[c], o[c]);
  }

  // Normalize and store attn output as [b][s][h*D + d] bf16.
#pragma unroll
  for (int c = 0; c < 4; ++c)
#pragma unroll
    for (int i = 0; i < 8; ++i) {
      const int s = q0 + i + 8 * half;
      const float val = o[c][i] / lrow[i];
      attnO[((long)(b * SS + s)) * DIMM + h * HD + c * 16 + ln] = f2bf(val);
    }
}

// ---------------------------------------------------------------------------
// Kernel 4: out = attn @ Wo + bo.  M=8192, N=1024, K=1024.  fp32 output.
// ---------------------------------------------------------------------------
__global__ __launch_bounds__(256) void out_gemm_kernel(
    const unsigned short* __restrict__ Ab, const unsigned short* __restrict__ WT,
    const float* __restrict__ bo, float* __restrict__ out) {
  const int wave = threadIdx.x >> 5, lane = threadIdx.x & 31;
  const int half = lane >> 4, ln = lane & 15;
  const int mBase = (blockIdx.x * 8 + wave) * 32;
  const int nBase = blockIdx.y * 64;
  v8f acc[8];
  gemm_core(acc, Ab, WT, mBase, nBase, half, ln);

#pragma unroll
  for (int sub = 0; sub < 2; ++sub)
#pragma unroll
    for (int c = 0; c < 4; ++c)
#pragma unroll
      for (int i = 0; i < 8; ++i) {
        const int m = mBase + sub * 16 + i + 8 * half;
        const int n = nBase + c * 16 + ln;
        out[(long)m * DIMM + n] = acc[sub * 4 + c][i] + bo[n];
      }
}

// ---------------------------------------------------------------------------
extern "C" void kernel_launch(void* const* d_in, const int* in_sizes, int n_in,
                              void* d_out, int out_size, void* d_ws, size_t ws_size,
                              hipStream_t stream) {
  const float* x  = (const float*)d_in[0];
  const float* Wq = (const float*)d_in[1];
  const float* bq = (const float*)d_in[2];
  const float* Wk = (const float*)d_in[3];
  const float* bk = (const float*)d_in[4];
  const float* Wv = (const float*)d_in[5];
  const float* bv = (const float*)d_in[6];
  const float* Wo = (const float*)d_in[7];
  const float* bo = (const float*)d_in[8];
  float* out = (float*)d_out;

  char* ws = (char*)d_ws;
  size_t off = 0;
  auto take = [&](size_t bytes) -> char* {
    char* p = ws + off;
    off = (off + bytes + 255) & ~(size_t)255;
    return p;
  };
  unsigned short* xb    = (unsigned short*)take((size_t)BB * SS * DIMM * 2);
  unsigned short* WqkvT = (unsigned short*)take((size_t)1280 * DIMM * 2);
  unsigned short* WoT   = (unsigned short*)take((size_t)DIMM * DIMM * 2);
  float*          biasC = (float*)take(1280 * 4);
  unsigned short* Qb    = (unsigned short*)take((size_t)BB * HH * SS * HD * 2);
  unsigned short* Kb    = (unsigned short*)take((size_t)BB * KVH * SS * HD * 2);
  unsigned short* Vtb   = (unsigned short*)take((size_t)BB * KVH * HD * SS * 2);
  unsigned short* attnO = (unsigned short*)take((size_t)BB * SS * DIMM * 2);
  (void)in_sizes; (void)n_in; (void)out_size; (void)ws_size;

  pack_kernel<<<2048, 256, 0, stream>>>(x, Wq, bq, Wk, bk, Wv, bv, Wo,
                                        xb, WqkvT, WoT, biasC);
  qkv_gemm_kernel<<<dim3(32, 20), 256, 0, stream>>>(xb, WqkvT, biasC, Qb, Kb, Vtb);
  attn_kernel<<<1024, 256, 0, stream>>>(Qb, Kb, Vtb, attnO);
  out_gemm_kernel<<<dim3(32, 16), 256, 0, stream>>>(attnO, WoT, bo, out);
}
